// NeuralFingerprint_75634374082560
// MI455X (gfx1250) — matrix operand order repaired and checked by reading, not compiled, not run on hardware
//
#include <hip/hip_runtime.h>
#include <cstdint>

#define NNODES 100000
#define NEDGES 1600000
#define FDIM   128
#define PDIM   2048
#define RADIUS 4

typedef __attribute__((ext_vector_type(16))) __bf16          v16bf;
typedef __attribute__((ext_vector_type(16))) unsigned short  v16us;
typedef __attribute__((ext_vector_type(8)))  float           v8f;

union BF16Frag { v16us u; v16bf b; };

__device__ __forceinline__ unsigned short f32_to_bf16(float f) {
  union { float f; unsigned int u; } v; v.f = f;
  unsigned int r = v.u + 0x7FFFu + ((v.u >> 16) & 1u);   // round-to-nearest-even
  return (unsigned short)(r >> 16);
}

// ---------------------------------------------------------------------------
// Pre-swizzle a row-major fp32 weight matrix [FDIM x Ncols] into bf16 stored
// in exact B-fragment order: frag index = ((ntile*4 + kt)*32 + lane), 16
// halves per lane with K = i + 16*(lane>>4), N = ntile*16 + (lane&15).
// Each B fragment then loads as a single contiguous 32B v16us.
// ---------------------------------------------------------------------------
__global__ void swizzle_w_kernel(const float* __restrict__ W,
                                 unsigned short* __restrict__ out, int Ncols) {
  int idx = blockIdx.x * blockDim.x + threadIdx.x;
  if (idx >= FDIM * Ncols) return;
  int i    = idx & 15;
  int lane = (idx >> 4) & 31;
  int kt   = (idx >> 9) & 3;
  int nt   = idx >> 11;
  int k = kt * 32 + i + 16 * (lane >> 4);
  int n = nt * 16 + (lane & 15);
  out[idx] = f32_to_bf16(W[k * Ncols + n]);
}

// ---------------------------------------------------------------------------
// neigh[dst] += x[src] : one 128-lane slice per edge, coalesced row traffic,
// float atomics into the destination row.
// ---------------------------------------------------------------------------
__global__ void scatter_kernel(const float* __restrict__ x,
                               const int* __restrict__ src,
                               const int* __restrict__ dst,
                               float* __restrict__ neigh) {
  long long idx = (long long)blockIdx.x * blockDim.x + threadIdx.x;
  int e = (int)(idx >> 7);
  int f = (int)(idx & 127);
  if (e >= NEDGES) return;
  int s = src[e];
  int d = dst[e];
  atomicAdd(&neigh[(long long)d * FDIM + f], x[(long long)s * FDIM + f]);
}

// Build an A fragment (16x32 bf16, MxK) from an LDS bf16 tile, in contiguous k order (same as the B swizzle):
// lane<16: M=lane, K = i; lane>=16 adds +16 to K.
__device__ __forceinline__ void build_a_frags(const unsigned short (*sA)[132],
                                              int lane, BF16Frag afrag[4]) {
  int half = lane >> 4;
  int m    = lane & 15;
#pragma unroll
  for (int kt = 0; kt < 4; ++kt) {
#pragma unroll
    for (int i = 0; i < 16; ++i) {
      int k = kt * 32 + i + 16 * half;
      afrag[kt].u[i] = sA[m][k];
    }
  }
}

// ---------------------------------------------------------------------------
// h = relu((x + neigh) @ W1 + b1).  256 threads = 8 waves; block owns 16 rows,
// each wave owns one 16-col tile; K=128 -> 4 bf16 WMMAs per wave.
// ---------------------------------------------------------------------------
__global__ void lin1_kernel(const float* __restrict__ x,
                            const float* __restrict__ neigh,
                            const unsigned short* __restrict__ W1s,
                            const float* __restrict__ b1,
                            float* __restrict__ h) {
  __shared__ __align__(16) unsigned short sA[16][132];
  int tid = threadIdx.x;
  int rowbase = blockIdx.x * 16;

  for (int t = tid; t < 16 * FDIM; t += 256) {
    int r = t >> 7, c = t & 127;
    long long g = (long long)(rowbase + r) * FDIM + c;
    sA[r][c] = f32_to_bf16(x[g] + neigh[g]);
  }
  __syncthreads();

  int wave = tid >> 5;          // column tile 0..7
  int lane = tid & 31;
  int half = lane >> 4, m = lane & 15;

  BF16Frag afrag[4];
  build_a_frags(sA, lane, afrag);

  v8f acc = {};
#pragma unroll
  for (int kt = 0; kt < 4; ++kt) {
    BF16Frag bf;
    bf.u = *(const v16us*)(W1s + (size_t)(((wave * 4) + kt) * 32 + lane) * 16);
    acc = __builtin_amdgcn_wmma_f32_16x16x32_bf16(
        false, afrag[kt].b, false, bf.b, (short)0, acc, false, false);
  }

  int col = wave * 16 + m;
  float bias = b1[col];
#pragma unroll
  for (int r = 0; r < 8; ++r) {
    float v = acc[r] + bias;
    v = v > 0.f ? v : 0.f;
    h[(long long)(rowbase + r + 8 * half) * FDIM + col] = v;
  }
}

// ---------------------------------------------------------------------------
// logits = h @ W2 + b2 ; fp += column-sums of softmax(logits).
// 512 threads = 16 waves; block owns 16 rows x 2048 cols; each wave owns 8
// column tiles (all logits stay in registers -> single-pass softmax).
// ---------------------------------------------------------------------------
__global__ void lin2_kernel(const float* __restrict__ h,
                            const unsigned short* __restrict__ W2s,
                            const float* __restrict__ b2,
                            float* __restrict__ fp) {
  __shared__ __align__(16) unsigned short sA[16][132];
  __shared__ float redbuf[16][16];   // [wave][row]
  __shared__ float grow[16];         // per-row global max, then denom

  int tid = threadIdx.x;
  int rowbase = blockIdx.x * 16;

  for (int t = tid; t < 16 * FDIM; t += 512) {
    int r = t >> 7, c = t & 127;
    sA[r][c] = f32_to_bf16(h[(long long)(rowbase + r) * FDIM + c]);
  }
  __syncthreads();

  int wave = tid >> 5;          // 0..15
  int lane = tid & 31;
  int half = lane >> 4, m = lane & 15;
  int ntbase = wave * 8;        // first of 8 column tiles for this wave

  BF16Frag afrag[4];
  build_a_frags(sA, lane, afrag);

  v8f acc[8];
#pragma unroll
  for (int ct = 0; ct < 8; ++ct) {
    v8f a = {};
#pragma unroll
    for (int kt = 0; kt < 4; ++kt) {
      BF16Frag bf;
      bf.u = *(const v16us*)(W2s + (size_t)((((ntbase + ct) * 4) + kt) * 32 + lane) * 16);
      a = __builtin_amdgcn_wmma_f32_16x16x32_bf16(
          false, afrag[kt].b, false, bf.b, (short)0, a, false, false);
    }
    float bias = b2[(ntbase + ct) * 16 + m];
#pragma unroll
    for (int r = 0; r < 8; ++r) a[r] += bias;
    acc[ct] = a;
  }

  // ---- per-row max over this wave's 128 columns, xor-shuffle over 16 lanes
  float rmax[8];
#pragma unroll
  for (int r = 0; r < 8; ++r) {
    float v = acc[0][r];
#pragma unroll
    for (int ct = 1; ct < 8; ++ct) v = fmaxf(v, acc[ct][r]);
#pragma unroll
    for (int d = 1; d < 16; d <<= 1) v = fmaxf(v, __shfl_xor(v, d, 32));
    rmax[r] = v;
  }
  if (m == 0) {
#pragma unroll
    for (int r = 0; r < 8; ++r) redbuf[wave][r + 8 * half] = rmax[r];
  }
  __syncthreads();
  if (tid < 16) {
    float v = redbuf[0][tid];
    for (int w = 1; w < 16; ++w) v = fmaxf(v, redbuf[w][tid]);
    grow[tid] = v;
  }
  __syncthreads();
  float gm[8];
#pragma unroll
  for (int r = 0; r < 8; ++r) gm[r] = grow[r + 8 * half];

  // ---- exponentiate in place + per-row partial sums
  float rsum[8];
#pragma unroll
  for (int r = 0; r < 8; ++r) rsum[r] = 0.f;
#pragma unroll
  for (int ct = 0; ct < 8; ++ct) {
#pragma unroll
    for (int r = 0; r < 8; ++r) {
      float e = __expf(acc[ct][r] - gm[r]);
      acc[ct][r] = e;
      rsum[r] += e;
    }
  }
#pragma unroll
  for (int r = 0; r < 8; ++r) {
#pragma unroll
    for (int d = 1; d < 16; d <<= 1) rsum[r] += __shfl_xor(rsum[r], d, 32);
  }
  __syncthreads();                       // redbuf free for reuse
  if (m == 0) {
#pragma unroll
    for (int r = 0; r < 8; ++r) redbuf[wave][r + 8 * half] = rsum[r];
  }
  __syncthreads();
  if (tid < 16) {
    float v = 0.f;
    for (int w = 0; w < 16; ++w) v += redbuf[w][tid];
    grow[tid] = v;
  }
  __syncthreads();
  float den[8];
#pragma unroll
  for (int r = 0; r < 8; ++r) den[r] = grow[r + 8 * half];

  // ---- column sums of softmax into fp (lane l ^ 16 holds the other 8 rows
  //      of the SAME column), then one atomic per column per block.
#pragma unroll
  for (int ct = 0; ct < 8; ++ct) {
    float s = 0.f;
#pragma unroll
    for (int r = 0; r < 8; ++r) s += acc[ct][r] / den[r];
    s += __shfl_xor(s, 16, 32);
    if (half == 0) atomicAdd(&fp[(ntbase + ct) * 16 + m], s);
  }
}

// ---------------------------------------------------------------------------
extern "C" void kernel_launch(void* const* d_in, const int* in_sizes, int n_in,
                              void* d_out, int out_size, void* d_ws, size_t ws_size,
                              hipStream_t stream) {
  const float* atoms = (const float*)d_in[0];
  const float* W1    = (const float*)d_in[1];
  const float* b1    = (const float*)d_in[2];
  const float* W2    = (const float*)d_in[3];
  const float* b2    = (const float*)d_in[4];
  const int*   esrc  = (const int*)d_in[5];
  const int*   edst  = (const int*)d_in[6];
  float* fp = (float*)d_out;

  // workspace layout
  float* neigh = (float*)d_ws;                                  // N*F f32
  float* hA    = neigh + (size_t)NNODES * FDIM;                 // N*F f32
  float* hB    = hA    + (size_t)NNODES * FDIM;                 // N*F f32
  unsigned short* W1s = (unsigned short*)(hB + (size_t)NNODES * FDIM); // 128*128 bf16
  unsigned short* W2s = W1s + (size_t)FDIM * FDIM;                     // 128*2048 bf16

  hipMemsetAsync(d_out, 0, (size_t)PDIM * sizeof(float), stream);
  swizzle_w_kernel<<<(FDIM * FDIM + 255) / 256, 256, 0, stream>>>(W1, W1s, FDIM);
  swizzle_w_kernel<<<(FDIM * PDIM + 255) / 256, 256, 0, stream>>>(W2, W2s, PDIM);

  const float* x = atoms;
  float* hcur = hA;
  float* halt = hB;
  for (int r = 0; r < RADIUS; ++r) {
    hipMemsetAsync(neigh, 0, (size_t)NNODES * FDIM * sizeof(float), stream);
    long long tot = (long long)NEDGES * FDIM;
    scatter_kernel<<<(int)((tot + 255) / 256), 256, 0, stream>>>(x, esrc, edst, neigh);
    lin1_kernel<<<NNODES / 16, 256, 0, stream>>>(x, neigh, W1s, b1, hcur);
    lin2_kernel<<<NNODES / 16, 512, 0, stream>>>(hcur, W2s, b2, fp);
    x = hcur;
    float* t = hcur; hcur = halt; halt = t;
  }
}
